// MainModel_49546742726820
// MI455X (gfx1250) — compile-verified, matrix operand order checked by reading, not run on hardware
//
#include <hip/hip_runtime.h>

typedef __attribute__((ext_vector_type(16))) _Float16 v16h;
typedef __attribute__((ext_vector_type(8)))  _Float16 v8h;
typedef __attribute__((ext_vector_type(8)))  float    v8f;

#define BB    64
#define NF    45
#define NP    180
#define PR    192      // NP padded to 12 row-tiles of 16
#define DM    256
#define NH    4
#define HD    64
#define DFFN  2048
#define HID   128
#define NCLS  10
#define BF    (BB*NF)  // 2880

// ---------------------------------------------------------------------------
// WMMA helpers (gfx1250 wave32, v_wmma_f32_16x16x32_f16)
// ---------------------------------------------------------------------------
__device__ __forceinline__ v16h ld_frag(const _Float16* p) {
  // lane holds k [kb..kb+7] and [kb+16..kb+23] of its row (ISA VGPR layout)
  v8h lo = *(const v8h*)p;
  v8h hi = *(const v8h*)(p + 16);
  v16h r;
#pragma unroll
  for (int i = 0; i < 8; ++i) { r[i] = lo[i]; r[i + 8] = hi[i]; }
  return r;
}

// One 16x16 output tile per wave, looped over tiles of an (M=PR) x N GEMM with
// A in LDS (row-major, lda halves) and B^T in global (N x K row-major).
// RELU and OUT_HALF are compile-time so the epilogue is branch-free.
template <bool RELU, bool OUT_HALF>
__device__ __forceinline__ void gemm_stage(
    const _Float16* __restrict__ A, int lda,
    const _Float16* __restrict__ BT, int ldb,
    const float* __restrict__ bias,
    void* __restrict__ Cout, int ldc,
    int mt16, int nt16, int ksteps, int wv, int lane)
{
  const int ntiles = mt16 * nt16;
  for (int t = wv; t < ntiles; t += 8) {
    int mt = t % mt16, nt = t / mt16;
    v8f acc = {};
    const _Float16* ap = A  + (size_t)(mt * 16 + (lane & 15)) * lda + ((lane >> 4) << 3);
    const _Float16* bp = BT + (size_t)(nt * 16 + (lane & 15)) * ldb + ((lane >> 4) << 3);
    for (int ks = 0; ks < ksteps; ++ks) {
      acc = __builtin_amdgcn_wmma_f32_16x16x32_f16(
          false, ld_frag(ap), false, ld_frag(bp), (short)0, acc, false, false);
      ap += 32; bp += 32;
    }
    const int   n     = nt * 16 + (lane & 15);
    const float bv    = bias[n];
    const int   mbase = mt * 16 + ((lane >> 4) << 3);
#pragma unroll
    for (int r = 0; r < 8; ++r) {
      float v = acc[r] + bv;
      if (RELU) v = fmaxf(v, 0.f);
      if (OUT_HALF) ((_Float16*)Cout)[(mbase + r) * ldc + n] = (_Float16)v;
      else          ((float*)Cout)[(mbase + r) * ldc + n] = v;
    }
  }
}

// ---------------------------------------------------------------------------
// K0: per-sample ymin reduction
// ---------------------------------------------------------------------------
__global__ void k_ymin(const float* __restrict__ pts, float* __restrict__ ymin) {
  int b = blockIdx.x, tid = threadIdx.x;
  __shared__ float red[256];
  const int n = NF * NP;
  float m = 3.4e38f;
  for (int i = tid; i < n; i += 256)
    m = fminf(m, pts[((size_t)b * n + i) * 4 + 1]);
  red[tid] = m; __syncthreads();
  for (int s = 128; s > 0; s >>= 1) { if (tid < s) red[tid] = fminf(red[tid], red[tid + s]); __syncthreads(); }
  if (tid == 0) ymin[b] = red[0];
}

// ---------------------------------------------------------------------------
// K1: weight prep — f32 (K x N) -> f16 transposed (N x Kpad); Kpad = 1<<kpLog2
// ---------------------------------------------------------------------------
__global__ void k_prep(const float* __restrict__ w, _Float16* __restrict__ wT,
                       int K, int N, int kpLog2) {
  int i = blockIdx.x * 256 + threadIdx.x;
  if (i >= (N << kpLog2)) return;
  int n = i >> kpLog2, k = i & ((1 << kpLog2) - 1);
  wT[i] = (_Float16)(k < K ? w[k * N + n] : 0.f);
}

// ---------------------------------------------------------------------------
// K2: fused per-frame pipeline: KNN + aug -> MLP(16,64,128,256) -> attn pool
// One workgroup (8 waves) per (b,f). All activations stay in LDS (~190KB).
// ---------------------------------------------------------------------------
__global__ void k_pointpipe(
    const float* __restrict__ points, const float* __restrict__ ymin,
    const _Float16* __restrict__ w1T, const float* __restrict__ c1b,
    const _Float16* __restrict__ w2T, const float* __restrict__ c2b,
    const _Float16* __restrict__ w3T, const float* __restrict__ c3b,
    const _Float16* __restrict__ a1T, const float* __restrict__ a1b,
    const float* __restrict__ a2w,    const float* __restrict__ a2b,
    float* __restrict__ featf, _Float16* __restrict__ feath)
{
  extern __shared__ char smem[];
  float*    s_cur  = (float*)smem;                 // 180*4 f32     (2880 B)
  float*    s_left = (float*)(smem + 2880);        // 180*4 f32     (2880 B)
  _Float16* s_aug  = (_Float16*)(smem + 5760);     // 192*32 f16   (12288 B)
  _Float16* s_x1   = (_Float16*)(smem + 18048);    // 192*64 f16   (24576 B)
  _Float16* s_x2   = (_Float16*)(smem + 42624);    // 192*128 f16  (49152 B)
  _Float16* s_x3   = (_Float16*)(smem + 91776);    // 192*256 f16  (98304 B)
  float*    s_att  = (float*)(smem + 190080);      // 192 f32        (768 B)
  float*    s_T    = (float*)(smem + 42624);       // reuse x2: 192*64 f32

  const int bf = blockIdx.x, b = bf / NF, f = bf % NF;
  const int tid = threadIdx.x, lane = tid & 31, wv = tid >> 5;
  const float ym = ymin[b];
  const float* cur = points + (size_t)bf * NP * 4;

  for (int i = tid; i < NP * 4; i += 256) {
    float v = cur[i];
    if ((i & 3) == 1) v -= ym;
    s_cur[i] = v;
    float lv = 0.f;
    if (f > 0) { lv = cur[i - NP * 4]; if ((i & 3) == 1) lv -= ym; }
    s_left[i] = lv;
  }
  __syncthreads();

  // KNN (top-3 by squared xyz distance into previous frame) + aug row build
  if (tid < PR) {
    _Float16* row = s_aug + tid * 32;
    if (tid < NP) {
      float px = s_cur[tid * 4], py = s_cur[tid * 4 + 1], pz = s_cur[tid * 4 + 2];
      float b0 = 3.4e38f, b1 = 3.4e38f, b2 = 3.4e38f; int i0 = 0, i1 = 0, i2 = 0;
      for (int q = 0; q < NP; ++q) {
        float dx = px - s_left[q * 4], dy = py - s_left[q * 4 + 1], dz = pz - s_left[q * 4 + 2];
        float d = dx * dx + dy * dy + dz * dz;
        if      (d < b0) { b2 = b1; i2 = i1; b1 = b0; i1 = i0; b0 = d; i0 = q; }
        else if (d < b1) { b2 = b1; i2 = i1; b1 = d; i1 = q; }
        else if (d < b2) { b2 = d; i2 = q; }
      }
      int ids[3] = { i0, i1, i2 };
#pragma unroll
      for (int c = 0; c < 4; ++c) row[c] = (_Float16)s_cur[tid * 4 + c];
      for (int kk = 0; kk < 3; ++kk)
#pragma unroll
        for (int c = 0; c < 4; ++c)
          row[4 + kk * 4 + c] = (_Float16)(s_cur[tid * 4 + c] - s_left[ids[kk] * 4 + c]);
      for (int j = 16; j < 32; ++j) row[j] = (_Float16)0.f;
    } else {
      for (int j = 0; j < 32; ++j) row[j] = (_Float16)0.f;
    }
  }
  __syncthreads();

  // 16->64 (K padded to 32), 64->128, 128->256, 256->64 (attention logits)
  gemm_stage<true,  true >(s_aug, 32,  w1T, 32,  c1b, s_x1, 64,  12, 4,  1, wv, lane);
  __syncthreads();
  gemm_stage<true,  true >(s_x1, 64,   w2T, 64,  c2b, s_x2, 128, 12, 8,  2, wv, lane);
  __syncthreads();
  gemm_stage<true,  true >(s_x2, 128,  w3T, 128, c3b, s_x3, 256, 12, 16, 4, wv, lane);
  __syncthreads();
  gemm_stage<false, false>(s_x3, 256,  a1T, 256, a1b, s_T,  64,  12, 4,  8, wv, lane);
  __syncthreads();

  // softmax over the 64 channels, dot with a2w -> per-point scalar attention
  if (tid < PR) {
    float att = 0.f;
    if (tid < NP) {
      const float* tr = s_T + tid * 64;
      float mx = -3.4e38f;
      for (int j = 0; j < 64; ++j) mx = fmaxf(mx, tr[j]);
      float sum = 0.f, acc = 0.f;
      for (int j = 0; j < 64; ++j) {
        float e = __expf(tr[j] - mx);
        sum += e; acc += e * a2w[j];
      }
      att = acc / sum + a2b[0];
    }
    s_att[tid] = att;
  }
  __syncthreads();

  // feat[c] = sum_p att[p] * x3[p][c]
  {
    float acc = 0.f;
    for (int p = 0; p < NP; ++p) acc += s_att[p] * (float)s_x3[p * 256 + tid];
    featf[(size_t)bf * DM + tid] = acc;
    feath[(size_t)bf * DM + tid] = (_Float16)acc;
  }
}

// ---------------------------------------------------------------------------
// K3: generic WMMA GEMM: C[M,N] = act(A[M,K] @ B + bias), B given as B^T (NxK)
// one 16x16 tile per wave, 4 waves / block; branch-free epilogue via templates
// ---------------------------------------------------------------------------
template <bool RELU, bool OUT_HALF>
__global__ void k_gemm(const _Float16* __restrict__ A, const _Float16* __restrict__ BT,
                       const float* __restrict__ bias, void* __restrict__ Cout,
                       int M, int N, int Kd)
{
  const int lane = threadIdx.x & 31, wv = threadIdx.x >> 5;
  const int mt16 = M >> 4, nt16 = N >> 4;
  const int tile = blockIdx.x * 4 + wv;
  if (tile >= mt16 * nt16) return;
  const int mt = tile % mt16, nt = tile / mt16;
  v8f acc = {};
  const _Float16* ap = A  + (size_t)(mt * 16 + (lane & 15)) * Kd + ((lane >> 4) << 3);
  const _Float16* bp = BT + (size_t)(nt * 16 + (lane & 15)) * Kd + ((lane >> 4) << 3);
  const int ksteps = Kd >> 5;
  for (int ks = 0; ks < ksteps; ++ks) {
    __builtin_prefetch(bp + 64, 0, 1);   // global_prefetch on weight stream
    acc = __builtin_amdgcn_wmma_f32_16x16x32_f16(
        false, ld_frag(ap), false, ld_frag(bp), (short)0, acc, false, false);
    ap += 32; bp += 32;
  }
  const int n = nt * 16 + (lane & 15);
  const float bv = bias[n];
  const int mbase = mt * 16 + ((lane >> 4) << 3);
#pragma unroll
  for (int r = 0; r < 8; ++r) {
    float v = acc[r] + bv;
    if (RELU) v = fmaxf(v, 0.f);
    if (OUT_HALF) ((_Float16*)Cout)[(size_t)(mbase + r) * N + n] = (_Float16)v;
    else          ((float*)Cout)[(size_t)(mbase + r) * N + n] = v;
  }
}

// ---------------------------------------------------------------------------
// K4: per-(b,head) scaled-dot attention on 45x64 tiles (tiny -> VALU)
// ---------------------------------------------------------------------------
__global__ void k_attn(const float* __restrict__ Q, const float* __restrict__ Km,
                       const float* __restrict__ V, _Float16* __restrict__ moh)
{
  const int b = blockIdx.x / NH, h = blockIdx.x % NH, tid = threadIdx.x; // 64 thr
  __shared__ float q[NF * HD], k[NF * HD], v[NF * HD], sc[NF * NF];
  for (int i = tid; i < NF * HD; i += 64) {
    int t = i / HD, d = i % HD;
    size_t gi = ((size_t)(b * NF + t)) * DM + h * HD + d;
    q[i] = Q[gi]; k[i] = Km[gi]; v[i] = V[gi];
  }
  __syncthreads();
  for (int i = tid; i < NF * NF; i += 64) {
    int t = i / NF, s = i % NF;
    float acc = 0.f;
    for (int d = 0; d < HD; ++d) acc += q[t * HD + d] * k[s * HD + d];
    sc[i] = acc * 0.125f;   // 1/sqrt(64)
  }
  __syncthreads();
  if (tid < NF) {
    float mx = -3.4e38f;
    for (int s = 0; s < NF; ++s) mx = fmaxf(mx, sc[tid * NF + s]);
    float sum = 0.f;
    for (int s = 0; s < NF; ++s) { float e = __expf(sc[tid * NF + s] - mx); sc[tid * NF + s] = e; sum += e; }
    float inv = 1.f / sum;
    for (int s = 0; s < NF; ++s) sc[tid * NF + s] *= inv;
  }
  __syncthreads();
  for (int i = tid; i < NF * HD; i += 64) {
    int t = i / HD, d = i % HD;
    float acc = 0.f;
    for (int s = 0; s < NF; ++s) acc += sc[t * NF + s] * v[s * HD + d];
    moh[((size_t)(b * NF + t)) * DM + h * HD + d] = (_Float16)acc;
  }
}

// ---------------------------------------------------------------------------
// K5: residual add + LayerNorm over DM=256, one block (256 thr) per row
// ---------------------------------------------------------------------------
template <bool OUTF>
__global__ void k_add_ln(const float* __restrict__ x1, const float* __restrict__ x2,
                         const float* __restrict__ g, const float* __restrict__ bt,
                         float* __restrict__ outf, _Float16* __restrict__ outh)
{
  const int row = blockIdx.x, c = threadIdx.x;
  __shared__ float red[256];
  float x = x1[(size_t)row * DM + c] + x2[(size_t)row * DM + c];
  red[c] = x; __syncthreads();
  for (int s = 128; s > 0; s >>= 1) { if (c < s) red[c] += red[c + s]; __syncthreads(); }
  float m = red[0] * (1.f / DM); __syncthreads();
  float d = x - m; red[c] = d * d; __syncthreads();
  for (int s = 128; s > 0; s >>= 1) { if (c < s) red[c] += red[c + s]; __syncthreads(); }
  float var = red[0] * (1.f / DM);
  float y = d * rsqrtf(var + 1e-5f) * g[c] + bt[c];
  if (OUTF) outf[(size_t)row * DM + c] = y;
  outh[(size_t)row * DM + c] = (_Float16)y;
}

// ---------------------------------------------------------------------------
// K6: BiLSTM recurrence (input GEMMs precomputed). One block per batch sample.
// ---------------------------------------------------------------------------
__global__ void k_lstm(const float* __restrict__ gxf, const float* __restrict__ gxb,
                       const float* __restrict__ whhf, const float* __restrict__ whhb,
                       float* __restrict__ hidden)
{
  const int b = blockIdx.x, tid = threadIdx.x; // 256 thr
  __shared__ float h[HID], c[HID], gg[4 * HID], hs[HID];
  if (tid < HID) hs[tid] = 0.f;
  for (int dir = 0; dir < 2; ++dir) {
    const float* gx  = dir ? gxb : gxf;
    const float* whh = dir ? whhb : whhf;
    if (tid < HID) { h[tid] = 0.f; c[tid] = 0.f; }
    __syncthreads();
    for (int step = 0; step < NF; ++step) {
      const int t = dir ? (NF - 1 - step) : step;
      const float* grow = gx + ((size_t)(b * NF + t)) * (4 * HID);
      for (int j = tid; j < 4 * HID; j += 256) {
        float acc = grow[j];
        for (int kk = 0; kk < HID; ++kk) acc += h[kk] * whh[kk * (4 * HID) + j];
        gg[j] = acc;
      }
      __syncthreads();
      if (tid < HID) {
        float ig = 1.f / (1.f + __expf(-gg[tid]));
        float fg = 1.f / (1.f + __expf(-gg[HID + tid]));
        float gc = tanhf(gg[2 * HID + tid]);
        float og = 1.f / (1.f + __expf(-gg[3 * HID + tid]));
        float cn = fg * c[tid] + ig * gc;
        c[tid] = cn;
        h[tid] = og * tanhf(cn);
      }
      __syncthreads();
    }
    if (tid < HID) hs[tid] += h[tid];
    __syncthreads();
  }
  if (tid < HID) hidden[b * HID + tid] = hs[tid];
}

// ---------------------------------------------------------------------------
// K7: classifier head + copy hidden to output
// ---------------------------------------------------------------------------
__global__ void k_predict(const float* __restrict__ hidden,
                          const float* __restrict__ p1w, const float* __restrict__ p1b,
                          const float* __restrict__ p2w, const float* __restrict__ p2b,
                          float* __restrict__ out)
{
  const int b = blockIdx.x, tid = threadIdx.x; // 64 thr
  __shared__ float z1[64];
  float acc = p1b[tid];
  for (int kk = 0; kk < HID; ++kk) acc += hidden[b * HID + kk] * p1w[kk * 64 + tid];
  z1[tid] = fmaxf(acc, 0.f);
  __syncthreads();
  if (tid < NCLS) {
    float a = p2b[tid];
    for (int j = 0; j < 64; ++j) a += z1[j] * p2w[j * NCLS + tid];
    out[b * NCLS + tid] = a;
  }
  for (int i = tid; i < HID; i += 64)
    out[BB * NCLS + b * HID + i] = hidden[b * HID + i];
}

// ---------------------------------------------------------------------------
// host side
// ---------------------------------------------------------------------------
extern "C" void kernel_launch(void* const* d_in, const int* in_sizes, int n_in,
                              void* d_out, int out_size, void* d_ws, size_t ws_size,
                              hipStream_t stream)
{
  const float* points = (const float*)d_in[0];
  const float* c1w = (const float*)d_in[1];  const float* c1b = (const float*)d_in[2];
  const float* c2w = (const float*)d_in[3];  const float* c2b = (const float*)d_in[4];
  const float* c3w = (const float*)d_in[5];  const float* c3b = (const float*)d_in[6];
  const float* a1w = (const float*)d_in[7];  const float* a1b = (const float*)d_in[8];
  const float* a2w = (const float*)d_in[9];  const float* a2b = (const float*)d_in[10];
  const float* wq  = (const float*)d_in[11]; const float* bq  = (const float*)d_in[12];
  const float* wk  = (const float*)d_in[13]; const float* bk  = (const float*)d_in[14];
  const float* wv  = (const float*)d_in[15]; const float* bv  = (const float*)d_in[16];
  const float* wo  = (const float*)d_in[17]; const float* bo  = (const float*)d_in[18];
  const float* ln1g = (const float*)d_in[19]; const float* ln1b = (const float*)d_in[20];
  const float* ff1w = (const float*)d_in[21]; const float* ff1b = (const float*)d_in[22];
  const float* ff2w = (const float*)d_in[23]; const float* ff2b = (const float*)d_in[24];
  const float* ln2g = (const float*)d_in[25]; const float* ln2b = (const float*)d_in[26];
  const float* wih_f = (const float*)d_in[27]; const float* whh_f = (const float*)d_in[28];
  const float* b_f   = (const float*)d_in[29];
  const float* wih_b = (const float*)d_in[30]; const float* whh_b = (const float*)d_in[31];
  const float* b_b   = (const float*)d_in[32];
  const float* p1w = (const float*)d_in[33]; const float* p1b = (const float*)d_in[34];
  const float* p2w = (const float*)d_in[35]; const float* p2b = (const float*)d_in[36];
  float* out = (float*)d_out;

  // ---- carve workspace ----
  char* base = (char*)d_ws;
  size_t off = 0;
  auto carve = [&](size_t bytes) { size_t o = off; off = (off + bytes + 255) & ~(size_t)255; return base + o; };

  float*     ymin  = (float*)carve(BB * 4);
  _Float16*  w1T   = (_Float16*)carve((size_t)64 * 32 * 2);
  _Float16*  w2T   = (_Float16*)carve((size_t)128 * 64 * 2);
  _Float16*  w3T   = (_Float16*)carve((size_t)256 * 128 * 2);
  _Float16*  a1T   = (_Float16*)carve((size_t)64 * 256 * 2);
  _Float16*  wqT   = (_Float16*)carve((size_t)256 * 256 * 2);
  _Float16*  wkT   = (_Float16*)carve((size_t)256 * 256 * 2);
  _Float16*  wvT   = (_Float16*)carve((size_t)256 * 256 * 2);
  _Float16*  woT   = (_Float16*)carve((size_t)256 * 256 * 2);
  _Float16*  ff1T  = (_Float16*)carve((size_t)DFFN * 256 * 2);
  _Float16*  ff2T  = (_Float16*)carve((size_t)256 * DFFN * 2);
  _Float16*  wihfT = (_Float16*)carve((size_t)512 * 256 * 2);
  _Float16*  wihbT = (_Float16*)carve((size_t)512 * 256 * 2);
  float*     featf = (float*)carve((size_t)BF * DM * 4);
  _Float16*  feath = (_Float16*)carve((size_t)BF * DM * 2);
  float*     Qf    = (float*)carve((size_t)BF * DM * 4);
  float*     Kf    = (float*)carve((size_t)BF * DM * 4);
  float*     Vf    = (float*)carve((size_t)BF * DM * 4);
  _Float16*  moh   = (_Float16*)carve((size_t)BF * DM * 2);
  float*     projf = (float*)carve((size_t)BF * DM * 4);
  float*     yf    = (float*)carve((size_t)BF * DM * 4);
  _Float16*  yh    = (_Float16*)carve((size_t)BF * DM * 2);
  _Float16*  h1h   = (_Float16*)carve((size_t)BF * DFFN * 2);
  float*     fff   = (float*)carve((size_t)BF * DM * 4);
  _Float16*  y2h   = (_Float16*)carve((size_t)BF * DM * 2);
  float*     gxf   = (float*)carve((size_t)BF * 512 * 4);
  float*     gxb   = (float*)carve((size_t)BF * 512 * 4);
  float*     hidden= (float*)carve((size_t)BB * HID * 4);

  // ---- K0: ymin ----
  k_ymin<<<BB, 256, 0, stream>>>(points, ymin);

  // ---- K1: weight prep (f32 KxN -> f16 NxKpad, Kpad = 1<<kpLog2) ----
  struct P { const float* w; _Float16* t; int K; int N; int kpLog2; };
  P preps[] = {
    { c1w,  w1T,   16,   64,   5  },
    { c2w,  w2T,   64,   128,  6  },
    { c3w,  w3T,   128,  256,  7  },
    { a1w,  a1T,   256,  64,   8  },
    { wq,   wqT,   256,  256,  8  },
    { wk,   wkT,   256,  256,  8  },
    { wv,   wvT,   256,  256,  8  },
    { wo,   woT,   256,  256,  8  },
    { ff1w, ff1T,  256,  DFFN, 8  },
    { ff2w, ff2T,  DFFN, 256,  11 },
    { wih_f,wihfT, 256,  512,  8  },
    { wih_b,wihbT, 256,  512,  8  },
  };
  for (auto& p : preps) {
    int total = p.N << p.kpLog2;
    k_prep<<<(total + 255) / 256, 256, 0, stream>>>(p.w, p.t, p.K, p.N, p.kpLog2);
  }

  // ---- K2: fused point pipeline (one WGP workgroup per frame) ----
  {
    const size_t smem = 190848;  // cur+left+aug+x1+x2+x3+att
    k_pointpipe<<<BF, 256, smem, stream>>>(points, ymin,
        w1T, c1b, w2T, c2b, w3T, c3b, a1T, a1b, a2w, a2b, featf, feath);
  }

  auto gemm_f = [&](const _Float16* A, const _Float16* BT, const float* bias,
                    float* Cf, int M, int N, int Kd) {
    int tiles = (M / 16) * (N / 16);
    k_gemm<false, false><<<(tiles + 3) / 4, 128, 0, stream>>>(A, BT, bias, Cf, M, N, Kd);
  };
  auto gemm_h_relu = [&](const _Float16* A, const _Float16* BT, const float* bias,
                         _Float16* Ch, int M, int N, int Kd) {
    int tiles = (M / 16) * (N / 16);
    k_gemm<true, true><<<(tiles + 3) / 4, 128, 0, stream>>>(A, BT, bias, Ch, M, N, Kd);
  };

  // ---- transformer layer ----
  gemm_f(feath, wqT, bq, Qf, BF, DM, DM);
  gemm_f(feath, wkT, bk, Kf, BF, DM, DM);
  gemm_f(feath, wvT, bv, Vf, BF, DM, DM);
  k_attn<<<BB * NH, 64, 0, stream>>>(Qf, Kf, Vf, moh);
  gemm_f(moh, woT, bo, projf, BF, DM, DM);
  k_add_ln<true><<<BF, 256, 0, stream>>>(featf, projf, ln1g, ln1b, yf, yh);
  gemm_h_relu(yh, ff1T, ff1b, h1h, BF, DFFN, DM);
  gemm_f(h1h, ff2T, ff2b, fff, BF, DM, DFFN);
  k_add_ln<false><<<BF, 256, 0, stream>>>(yf, fff, ln2g, ln2b, nullptr, y2h);

  // ---- BiLSTM ----
  gemm_f(y2h, wihfT, b_f, gxf, BF, 4 * HID, DM);
  gemm_f(y2h, wihbT, b_b, gxb, BF, 4 * HID, DM);
  k_lstm<<<BB, 256, 0, stream>>>(gxf, gxb, whh_f, whh_b, hidden);

  // ---- head + outputs (cl then hidden, concatenated) ----
  k_predict<<<BB, 64, 0, stream>>>(hidden, p1w, p1b, p2w, p2b, out);

  (void)in_sizes; (void)n_in; (void)out_size; (void)ws_size;
}
